// SlidingWindowAttention_19490561589550
// MI455X (gfx1250) — compile-verified, matrix operand order checked by reading, not run on hardware
//
#include <hip/hip_runtime.h>

typedef __attribute__((ext_vector_type(16))) __bf16    v16bf;
typedef __attribute__((ext_vector_type(8)))  __bf16    v8bf;
typedef __attribute__((ext_vector_type(8)))  float     v8f;
typedef __attribute__((ext_vector_type(4)))  unsigned  u32x4;
typedef __attribute__((ext_vector_type(8)))  unsigned  u32x8;

#define DEV __device__ __forceinline__

constexpr int   BATCH  = 2;
constexpr int   SEQ    = 2048;
constexpr int   DIM_C  = 2048;
constexpr int   NHEAD  = 16;
constexpr int   HDIM   = 128;
constexpr int   WIN    = 3072;
constexpr float SCALE  = 0.08838834764831845f;  // 1/sqrt(128)
constexpr float NEGINF = -1e9f;

// ---------------------------------------------------------------------------
// f32 -> bf16 conversion (4 elements per thread)
// ---------------------------------------------------------------------------
__global__ void cvt_bf16(const float* __restrict__ in, __bf16* __restrict__ out, int n) {
  int i = (blockIdx.x * blockDim.x + threadIdx.x) * 4;
  if (i + 3 < n) {
    float4 f = *(const float4*)(in + i);
    out[i + 0] = (__bf16)f.x;
    out[i + 1] = (__bf16)f.y;
    out[i + 2] = (__bf16)f.z;
    out[i + 3] = (__bf16)f.w;
  } else {
    for (; i < n; ++i) out[i] = (__bf16)in[i];
  }
}

// ---------------------------------------------------------------------------
// Load a 16x32 bf16 fragment in CDNA5 WMMA A/B layout from global memory.
// lane&15 = M row (A) / N column (B); (lane>>4)*8 = K sub-offset.
// Both 8-element halves are contiguous 16-byte loads.
// ---------------------------------------------------------------------------
DEV v16bf load_frag(const __bf16* base, int ld, int k0) {
  int lane = threadIdx.x & 31;
  int r    = lane & 15;
  int kb   = (lane >> 4) << 3;
  const __bf16* p = base + (size_t)r * ld + k0 + kb;
  v8bf lo = *(const v8bf*)p;          // K = k0+kb    .. +7
  v8bf hi = *(const v8bf*)(p + 16);   // K = k0+16+kb .. +7
  v16bf f;
#pragma unroll
  for (int i = 0; i < 8; ++i) { f[i] = lo[i]; f[i + 8] = hi[i]; }
  return f;
}

// ---------------------------------------------------------------------------
// GEMM: Y[m,n] = sum_k A[m,k] * W[n,k] + bias[n]   (i.e. Y = A @ W^T + b)
//
// The block's 16-row A strip (16 x K bf16 = 64 KB) is staged into LDS ONCE by
// a single Tensor Data Mover descriptor (tensor_load_to_lds, TENSORcnt), then
// all 8 waves re-read A fragments via ds_load_b128 while B streams from
// global.  Wave tile: 16x64 (4 WMMA accumulators).
//
// MODE 0: store bf16 row-major [M,NN]
// MODE 1: store bf16 transposed per batch: out[b][n][tok]  (for V^T)
// MODE 2: store f32 row-major [M,NN]  (final output projection)
// ---------------------------------------------------------------------------
template <int MODE>
__global__ __launch_bounds__(256) void gemm16(const __bf16* __restrict__ A,
                                              const __bf16* __restrict__ W,
                                              const float* __restrict__ bias,
                                              void* __restrict__ out,
                                              int M, int NN, int K) {
  __shared__ __bf16 sA[16 * DIM_C];  // 64 KB A strip for this block
  const int lane = threadIdx.x & 31;
  const int wave = threadIdx.x >> 5;
  const int m0   = blockIdx.x * 16;
  const int n0   = (blockIdx.y * 8 + wave) * 64;
  const __bf16* Abase = A + (size_t)m0 * K;

  // ---- TDM: DMA the 16 x K A strip into LDS (wave 0 only, then barrier) ----
  if (wave == 0) {
    unsigned long long ga = (unsigned long long)(size_t)Abase;
    unsigned lds = (unsigned)(unsigned long long)(size_t)&sA[0];  // LDS aperture: low 32 bits
    u32x4 g0;
    g0[0] = 1u;                                         // count=1, user descriptor
    g0[1] = lds;                                        // lds_addr
    g0[2] = (unsigned)(ga & 0xFFFFFFFFu);               // global_addr[31:0]
    g0[3] = (unsigned)((ga >> 32) & 0x1FFFFFFu)         // global_addr[56:32]
            | (2u << 30);                               // type = 2 (image)
    u32x8 g1;
    g1[0] = (1u << 16);                                 // data_size = 2 bytes
    g1[1] = ((unsigned)K & 0xFFFFu) << 16;              // tensor_dim0[15:0]
    g1[2] = (((unsigned)K >> 16) & 0xFFFFu)             // tensor_dim0[31:16]
            | (16u << 16);                              // tensor_dim1[15:0] = 16
    g1[3] = ((unsigned)K & 0xFFFFu) << 16;              // tile_dim0 = K (dim1 hi = 0)
    g1[4] = 16u;                                        // tile_dim1 = 16, tile_dim2 = 0
    g1[5] = (unsigned)K;                                // tensor_dim0_stride[31:0] = K
    g1[6] = 0u;                                         // stride hi / dim1_stride lo
    g1[7] = 0u;                                         // dim1_stride hi
    asm volatile("tensor_load_to_lds %0, %1" :: "s"(g0), "s"(g1) : "memory");
    __builtin_amdgcn_s_wait_tensorcnt(0);
  }
  __syncthreads();

  const int r   = lane & 15;
  const int kb  = (lane >> 4) << 3;

  v8f acc[4] = {};
  for (int k = 0; k < K; k += 32) {
    // A fragment from LDS (ds_load_b128 x2)
    v16bf a;
    {
      v8bf lo = *(const v8bf*)&sA[r * K + k + kb];
      v8bf hi = *(const v8bf*)&sA[r * K + k + 16 + kb];
#pragma unroll
      for (int i = 0; i < 8; ++i) { a[i] = lo[i]; a[i + 8] = hi[i]; }
    }
    if (k + 64 < K)  // warm next W tile (global_prefetch_b8)
      __builtin_prefetch(W + (size_t)(n0 + r) * K + k + 64, 0, 1);
#pragma unroll
    for (int t = 0; t < 4; ++t) {
      v16bf b = load_frag(W + (size_t)(n0 + 16 * t) * K, K, k);
      acc[t] = __builtin_amdgcn_wmma_f32_16x16x32_bf16(
          false, a, false, b, (short)0, acc[t], false, false);
    }
  }

  const int hi8 = (lane >> 4) * 8;  // C layout: lanes 16-31 hold rows M=8..15
#pragma unroll
  for (int t = 0; t < 4; ++t) {
    int   col = n0 + 16 * t + (lane & 15);
    float bv  = bias[col];
#pragma unroll
    for (int ii = 0; ii < 8; ++ii) {
      int   row = m0 + ii + hi8;
      float v   = acc[t][ii] + bv;
      if (MODE == 0) {
        ((__bf16*)out)[(size_t)row * NN + col] = (__bf16)v;
      } else if (MODE == 1) {
        int bb = row >> 11, tok = row & (SEQ - 1);
        ((__bf16*)out)[((size_t)bb * NN + col) * SEQ + tok] = (__bf16)v;
      } else {
        ((float*)out)[(size_t)row * NN + col] = v;
      }
    }
  }
}

// ---------------------------------------------------------------------------
// Sliding-window flash attention.  One wave owns a 16-row Q tile of one
// (batch, head); iterates 32-key tiles over the band [max(0, i-1024), i].
// Q/K in bf16 [B*N, DIM]; V pre-transposed bf16 [B][DIM][N]; output bf16
// [B*N, DIM].  Softmax row stats reduced across 16-lane halves (wave32).
// P is re-laid C-fragment -> A-fragment through per-wave LDS.
// Barrier-free (waves have divergent trip counts).
// ---------------------------------------------------------------------------
__global__ __launch_bounds__(256) void swa_attn(const __bf16* __restrict__ Q,
                                                const __bf16* __restrict__ Kt,
                                                const __bf16* __restrict__ Vt,
                                                __bf16* __restrict__ O) {
  __shared__ __bf16 sP[8][16 * 32];  // 1 KB per wave
  const int lane = threadIdx.x & 31;
  const int wv   = threadIdx.x >> 5;
  const int w    = blockIdx.x * 8 + wv;
  const int qt   = w & 127;          // 128 q-tiles per (b,h)
  const int h    = (w >> 7) & 15;
  const int b    = w >> 11;
  const int q0   = qt * 16;
  const int hi8  = (lane >> 4) * 8;

  // Q fragments for the full head dim (K = 128 -> 4 x 32)
  const __bf16* Qbase = Q + (size_t)(b * SEQ + q0) * DIM_C + h * HDIM;
  v16bf qf[4];
#pragma unroll
  for (int kk = 0; kk < 4; ++kk) qf[kk] = load_frag(Qbase, DIM_C, 32 * kk);

  v8f   acc[8] = {};
  float mrow[8], lrow[8];
#pragma unroll
  for (int ii = 0; ii < 8; ++ii) { mrow[ii] = -3.0e38f; lrow[ii] = 0.f; }

  int jbeg = q0 + (SEQ - WIN);       // q0 - 1024
  if (jbeg < 0) jbeg = 0;
  jbeg &= ~31;

  for (int j0 = jbeg; j0 < q0 + 16; j0 += 32) {
    // ---- S = Q K^T over two 16-column tiles ----
    v8f s[2];
#pragma unroll
    for (int t = 0; t < 2; ++t) {
      v8f st = {};
      const __bf16* Kb = Kt + (size_t)(b * SEQ + j0 + 16 * t) * DIM_C + h * HDIM;
#pragma unroll
      for (int kk = 0; kk < 4; ++kk) {
        v16bf kf = load_frag(Kb, DIM_C, 32 * kk);
        st = __builtin_amdgcn_wmma_f32_16x16x32_bf16(
            false, qf[kk], false, kf, (short)0, st, false, false);
      }
      s[t] = st;
    }

    // ---- band mask + scale ----
#pragma unroll
    for (int t = 0; t < 2; ++t) {
      int j = j0 + 16 * t + (lane & 15);
#pragma unroll
      for (int ii = 0; ii < 8; ++ii) {
        int  i  = q0 + ii + hi8;
        bool ok = (j <= i) && (j >= i + (SEQ - WIN));
        s[t][ii] = ok ? s[t][ii] * SCALE : NEGINF;
      }
    }

    // ---- online softmax (row reductions across 16-lane halves) ----
    float corr[8];
#pragma unroll
    for (int ii = 0; ii < 8; ++ii) {
      float mx = fmaxf(s[0][ii], s[1][ii]);
#pragma unroll
      for (int d = 8; d >= 1; d >>= 1) mx = fmaxf(mx, __shfl_xor(mx, d, 32));
      float mn = fmaxf(mrow[ii], mx);
      corr[ii] = __expf(mrow[ii] - mn);
      mrow[ii] = mn;
      float p0 = __expf(s[0][ii] - mn);
      float p1 = __expf(s[1][ii] - mn);
      s[0][ii] = p0;
      s[1][ii] = p1;
      float rs = p0 + p1;
#pragma unroll
      for (int d = 8; d >= 1; d >>= 1) rs += __shfl_xor(rs, d, 32);
      lrow[ii] = lrow[ii] * corr[ii] + rs;
    }
#pragma unroll
    for (int c = 0; c < 8; ++c)
#pragma unroll
      for (int ii = 0; ii < 8; ++ii) acc[c][ii] *= corr[ii];

    // ---- P: C-fragment -> row-major LDS (16x32 bf16) ----
#pragma unroll
    for (int t = 0; t < 2; ++t) {
      int col = 16 * t + (lane & 15);
#pragma unroll
      for (int ii = 0; ii < 8; ++ii) {
        int row = ii + hi8;
        sP[wv][row * 32 + col] = (__bf16)s[t][ii];
      }
    }
    asm volatile("s_wait_dscnt 0" ::: "memory");

    // ---- re-read P as A fragment ----
    v16bf pf;
    {
      int  rr = lane & 15;
      int  kb = (lane >> 4) << 3;
      v8bf lo = *(const v8bf*)&sP[wv][rr * 32 + kb];
      v8bf hi = *(const v8bf*)&sP[wv][rr * 32 + 16 + kb];
#pragma unroll
      for (int i = 0; i < 8; ++i) { pf[i] = lo[i]; pf[i + 8] = hi[i]; }
    }

    // ---- O += P V  (V^T rows are j-contiguous) ----
#pragma unroll
    for (int c = 0; c < 8; ++c) {
      const __bf16* Vb = Vt + ((size_t)b * DIM_C + h * HDIM + c * 16) * SEQ;
      v16bf vf = load_frag(Vb, SEQ, j0);
      acc[c] = __builtin_amdgcn_wmma_f32_16x16x32_bf16(
          false, pf, false, vf, (short)0, acc[c], false, false);
    }
  }

  // ---- normalize and write out ----
#pragma unroll
  for (int c = 0; c < 8; ++c) {
    int d = h * HDIM + c * 16 + (lane & 15);
#pragma unroll
    for (int ii = 0; ii < 8; ++ii) {
      int tok = q0 + ii + hi8;
      O[(size_t)(b * SEQ + tok) * DIM_C + d] = (__bf16)(acc[c][ii] / lrow[ii]);
    }
  }
}

// ---------------------------------------------------------------------------
// Launcher
// ---------------------------------------------------------------------------
extern "C" void kernel_launch(void* const* d_in, const int* in_sizes, int n_in,
                              void* d_out, int out_size, void* d_ws, size_t ws_size,
                              hipStream_t stream) {
  const float* x  = (const float*)d_in[0];
  const float* Wq = (const float*)d_in[1];
  const float* bq = (const float*)d_in[2];
  const float* Wk = (const float*)d_in[3];
  const float* bk = (const float*)d_in[4];
  const float* Wv = (const float*)d_in[5];
  const float* bv = (const float*)d_in[6];
  const float* Wo = (const float*)d_in[7];
  const float* bo = (const float*)d_in[8];

  char* ws = (char*)d_ws;
  const size_t MB = 1024 * 1024;
  __bf16* xb  = (__bf16*)(ws + 0);        // 16 MB  [B*N, DIM]
  __bf16* Wqb = (__bf16*)(ws + 16 * MB);  //  8 MB
  __bf16* Wkb = (__bf16*)(ws + 24 * MB);  //  8 MB
  __bf16* Wvb = (__bf16*)(ws + 32 * MB);  //  8 MB
  __bf16* Wob = (__bf16*)(ws + 40 * MB);  //  8 MB
  __bf16* Qb  = (__bf16*)(ws + 48 * MB);  // 16 MB  [B*N, DIM]
  __bf16* Kb  = (__bf16*)(ws + 64 * MB);  // 16 MB  [B*N, DIM]
  __bf16* Vtb = (__bf16*)(ws + 80 * MB);  // 16 MB  [B][DIM][N]
  __bf16* Ab  = (__bf16*)(ws + 96 * MB);  // 16 MB  [B*N, DIM]

  const int M  = BATCH * SEQ;   // 4096
  const int NN = DIM_C;         // 2048
  const int K  = DIM_C;         // 2048
  const int nx = M * DIM_C;     // 8388608
  const int nw = DIM_C * DIM_C; // 4194304

  cvt_bf16<<<(nx / 4 + 255) / 256, 256, 0, stream>>>(x, xb, nx);
  cvt_bf16<<<(nw / 4 + 255) / 256, 256, 0, stream>>>(Wq, Wqb, nw);
  cvt_bf16<<<(nw / 4 + 255) / 256, 256, 0, stream>>>(Wk, Wkb, nw);
  cvt_bf16<<<(nw / 4 + 255) / 256, 256, 0, stream>>>(Wv, Wvb, nw);
  cvt_bf16<<<(nw / 4 + 255) / 256, 256, 0, stream>>>(Wo, Wob, nw);

  dim3 gg(M / 16, NN / 512);  // (256, 4); 8 waves x 64 cols per block
  gemm16<0><<<gg, 256, 0, stream>>>(xb, Wqb, bq, Qb,  M, NN, K);
  gemm16<0><<<gg, 256, 0, stream>>>(xb, Wkb, bk, Kb,  M, NN, K);
  gemm16<1><<<gg, 256, 0, stream>>>(xb, Wvb, bv, Vtb, M, NN, K);

  swa_attn<<<BATCH * NHEAD * (SEQ / 16) / 8, 256, 0, stream>>>(Qb, Kb, Vtb, Ab);

  gemm16<2><<<gg, 256, 0, stream>>>(Ab, Wob, bo, d_out, M, NN, K);
}
